// Discriminator_15908558864540
// MI455X (gfx1250) — compile-verified, matrix operand order checked by reading, not run on hardware
//
#include <hip/hip_runtime.h>
#include <hip/hip_bf16.h>

typedef __attribute__((ext_vector_type(16))) __bf16 v16bf;
typedef __attribute__((ext_vector_type(8)))  __bf16 v8bf;
typedef __attribute__((ext_vector_type(8)))  float  v8f;

#define LEAK 0.2f

// ---------------------------------------------------------------------------
// f32 -> bf16 elementwise convert
__global__ void cvt_bf16_kernel(const float* __restrict__ src,
                                __bf16* __restrict__ dst, int n) {
    for (int i = blockIdx.x * blockDim.x + threadIdx.x; i < n;
         i += gridDim.x * blockDim.x) {
        dst[i] = (__bf16)src[i];
    }
}

// ---------------------------------------------------------------------------
// Transpose + convert: src is K x N row-major f32, dst is N x K row-major bf16
__global__ void transpose_cvt_kernel(const float* __restrict__ src,
                                     __bf16* __restrict__ dst, int K, int N) {
    int total = K * N;
    for (int i = blockIdx.x * blockDim.x + threadIdx.x; i < total;
         i += gridDim.x * blockDim.x) {
        int n = i / K;
        int k = i - n * K;
        dst[i] = (__bf16)src[(size_t)k * N + n];
    }
}

// ---------------------------------------------------------------------------
// Fragment load: two contiguous 16B loads -> one 16-element bf16 fragment.
// 16-bit A/B fragment layout (ISA 7.12.2): lane l holds row/col (l&15);
// lanes 0-15 carry K offsets {0..7, 16..23}, lanes 16-31 {8..15, 24..31}.
__device__ __forceinline__ v16bf load_frag(const __bf16* p) {
    v8bf lo = *(const v8bf*)p;
    v8bf hh = *(const v8bf*)(p + 16);
    return __builtin_shufflevector(lo, hh, 0, 1, 2, 3, 4, 5, 6, 7,
                                   8, 9, 10, 11, 12, 13, 14, 15);
}

#define WMMA_BF16(a, b, c)                                               \
    __builtin_amdgcn_wmma_f32_16x16x32_bf16(false, (a), false, (b),      \
                                            (short)0, (c), false, false)

// ---------------------------------------------------------------------------
// WMMA GEMM: out = leaky_relu(A @ B^T + bias)
//   A  : M x K bf16 row-major
//   Bt : N x K bf16 row-major (i.e. B transposed)
// One wave32 per 16x64 output tile (4 accumulators). The k-loop is manually
// software-pipelined: iteration kt issues the loads for kt+32 BEFORE the four
// v_wmma_f32_16x16x32_bf16 ops consuming kt's fragments, so the WMMAs only
// need s_wait_loadcnt <= (in-flight next-iter loads) instead of a full drain.
__global__ void wmma_gemm_bias_leaky(const __bf16* __restrict__ A,
                                     const __bf16* __restrict__ Bt,
                                     const float* __restrict__ bias,
                                     __bf16* __restrict__ outb,
                                     float*  __restrict__ outf,
                                     int M, int N, int K) {
    const int lane = threadIdx.x & 31;
    const int wave = threadIdx.x >> 5;
    const int tiles_n = N >> 6;               // 64-wide N tiles
    const int tile = blockIdx.x * (blockDim.x >> 5) + wave;
    if (tile >= (M >> 4) * tiles_n) return;   // wave-uniform

    const int tm = tile / tiles_n;
    const int tn = tile - tm * tiles_n;
    const int r  = lane & 15;
    const int hi = lane >> 4;

    const __bf16* ap  = A  + (size_t)(tm * 16 + r) * K + hi * 8;
    const __bf16* bp0 = Bt + (size_t)(tn * 64 + r) * K + hi * 8;
    const __bf16* bp1 = bp0 + (size_t)16 * K;
    const __bf16* bp2 = bp0 + (size_t)32 * K;
    const __bf16* bp3 = bp0 + (size_t)48 * K;

    v8f c0 = {}, c1 = {}, c2 = {}, c3 = {};

    // prologue: fragments for kt = 0
    v16bf a  = load_frag(ap);
    v16bf b0 = load_frag(bp0);
    v16bf b1 = load_frag(bp1);
    v16bf b2 = load_frag(bp2);
    v16bf b3 = load_frag(bp3);

    for (int kt = 32; kt < K; kt += 32) {
        ap += 32; bp0 += 32; bp1 += 32; bp2 += 32; bp3 += 32;
        // prefetch next k-step into the second buffer
        v16bf an  = load_frag(ap);
        v16bf bn0 = load_frag(bp0);
        v16bf bn1 = load_frag(bp1);
        v16bf bn2 = load_frag(bp2);
        v16bf bn3 = load_frag(bp3);
        // consume current fragments (loads above still in flight)
        c0 = WMMA_BF16(a, b0, c0);
        c1 = WMMA_BF16(a, b1, c1);
        c2 = WMMA_BF16(a, b2, c2);
        c3 = WMMA_BF16(a, b3, c3);
        a = an; b0 = bn0; b1 = bn1; b2 = bn2; b3 = bn3;
    }
    c0 = WMMA_BF16(a, b0, c0);
    c1 = WMMA_BF16(a, b1, c1);
    c2 = WMMA_BF16(a, b2, c2);
    c3 = WMMA_BF16(a, b3, c3);

    // C/D layout: VGPR v holds M = v + 8*hi, N = lane&15 (within subtile)
    const float* cs[4] = {(const float*)&c0, (const float*)&c1,
                          (const float*)&c2, (const float*)&c3};
#pragma unroll
    for (int t = 0; t < 4; ++t) {
        const int n = tn * 64 + t * 16 + r;
        const float bv = bias[n];
#pragma unroll
        for (int v = 0; v < 8; ++v) {
            const int m = tm * 16 + v + hi * 8;
            float val = cs[t][v] + bv;
            val = val > 0.0f ? val : val * LEAK;
            if (outf) outf[(size_t)m * N + n] = val;
            if (outb) outb[(size_t)m * N + n] = (__bf16)val;
        }
    }
}

// ---------------------------------------------------------------------------
// m = h2 @ T, written transposed: mt[f][i]  (f32, full precision)
// One block per batch row i; h2 row staged in LDS; T reads coalesced across f.
__global__ void mproj_kernel(const float* __restrict__ h2,
                             const float* __restrict__ T,
                             float* __restrict__ mt, int B, int Kc, int F) {
    __shared__ float sh[256];
    const int i = blockIdx.x;
    for (int k = threadIdx.x; k < Kc; k += blockDim.x)
        sh[k] = h2[(size_t)i * Kc + k];
    __syncthreads();
    const int f = threadIdx.x;
    if (f < F) {
        float acc = 0.0f;
#pragma unroll 4
        for (int k = 0; k < Kc; ++k)
            acc += sh[k] * T[(size_t)k * F + f];
        mt[(size_t)f * B + i] = acc;
    }
}

// ---------------------------------------------------------------------------
// feats[i][f] = sum_j exp(-|m[i][f] - m[j][f]|)
// blockIdx.x = feature f, blockIdx.y = chunk of 256 rows i.
// Entire column m[:,f] (8KB) staged in LDS, broadcast-read (conflict free).
__global__ void feats_kernel(const float* __restrict__ mt,
                             float* __restrict__ feats, int B, int F) {
    __shared__ float sm[2048];
    const int f = blockIdx.x;
    const float* col = mt + (size_t)f * B;
    for (int j = threadIdx.x; j < B; j += blockDim.x) sm[j] = col[j];
    __syncthreads();

    const int i = blockIdx.y * blockDim.x + threadIdx.x;
    const float mi = sm[i];
    float acc = 0.0f;
#pragma unroll 8
    for (int j = 0; j < B; ++j)
        acc += __expf(-fabsf(mi - sm[j]));
    feats[(size_t)i * F + f] = acc;
}

// ---------------------------------------------------------------------------
// out[i] = [h2_i ; feats_i] . Wf + bf
__global__ void final_kernel(const float* __restrict__ h2,
                             const float* __restrict__ feats,
                             const float* __restrict__ Wf,
                             const float* __restrict__ bf,
                             float* __restrict__ out, int B, int H, int F) {
    __shared__ float w[512];
    const int tot = H + F;
    for (int k = threadIdx.x; k < tot; k += blockDim.x) w[k] = Wf[k];
    __syncthreads();
    const int i = blockIdx.x * blockDim.x + threadIdx.x;
    if (i < B) {
        float acc = bf[0];
        for (int k = 0; k < H; ++k) acc += h2[(size_t)i * H + k] * w[k];
        for (int f = 0; f < F; ++f) acc += feats[(size_t)i * F + f] * w[H + f];
        out[i] = acc;
    }
}

// ---------------------------------------------------------------------------
extern "C" void kernel_launch(void* const* d_in, const int* in_sizes, int n_in,
                              void* d_out, int out_size, void* d_ws, size_t ws_size,
                              hipStream_t stream) {
    const float* x  = (const float*)d_in[0];
    const float* W1 = (const float*)d_in[1];
    const float* b1 = (const float*)d_in[2];
    const float* W2 = (const float*)d_in[3];
    const float* b2 = (const float*)d_in[4];
    const float* T  = (const float*)d_in[5];
    const float* Wf = (const float*)d_in[6];
    const float* bf = (const float*)d_in[7];

    const int H1 = in_sizes[2];        // 512
    const int H2 = in_sizes[4];        // 256
    const int D  = in_sizes[1] / H1;   // 3072
    const int B  = in_sizes[0] / D;    // 2048
    const int F  = in_sizes[5] / H2;   // 100

    // Workspace carve-out (256B aligned): ~22 MB total
    char* p = (char*)d_ws;
    auto carve = [&](size_t bytes) -> char* {
        char* r = p;
        p += (bytes + 255) & ~(size_t)255;
        return r;
    };
    __bf16* xb  = (__bf16*)carve((size_t)B * D * 2);
    __bf16* w1t = (__bf16*)carve((size_t)D * H1 * 2);
    __bf16* w2t = (__bf16*)carve((size_t)H1 * H2 * 2);
    __bf16* h1b = (__bf16*)carve((size_t)B * H1 * 2);
    float*  h2f = (float*)carve((size_t)B * H2 * 4);
    float*  mt  = (float*)carve((size_t)F * B * 4);
    float*  ft  = (float*)carve((size_t)B * F * 4);

    // Stage 0: precision conversion / weight transposition
    cvt_bf16_kernel<<<1024, 256, 0, stream>>>(x, xb, B * D);
    transpose_cvt_kernel<<<1024, 256, 0, stream>>>(W1, w1t, D, H1);
    transpose_cvt_kernel<<<256, 256, 0, stream>>>(W2, w2t, H1, H2);

    // Stage 1: h1 = leaky(x @ W1 + b1)   (bf16 out only; feeds GEMM2)
    const int tiles1 = (B / 16) * (H1 / 64);       // 16x64 tiles per wave
    wmma_gemm_bias_leaky<<<tiles1 / 8, 256, 0, stream>>>(
        xb, w1t, b1, h1b, (float*)nullptr, B, H1, D);

    // Stage 2: h2 = leaky(h1 @ W2 + b2)  (f32 out; needed exactly downstream)
    const int tiles2 = (B / 16) * (H2 / 64);
    wmma_gemm_bias_leaky<<<tiles2 / 8, 256, 0, stream>>>(
        h1b, w2t, b2, (__bf16*)nullptr, h2f, B, H2, H1);

    // Stage 3: mt[f][i] = (h2 @ T)^T
    mproj_kernel<<<B, 128, 0, stream>>>(h2f, T, mt, B, H2, F);

    // Stage 4: minibatch discrimination features
    feats_kernel<<<dim3(F, B / 256), 256, 0, stream>>>(mt, ft, B, F);

    // Stage 5: out = [h2; feats] @ Wf + bf
    final_kernel<<<(B + 255) / 256, 256, 0, stream>>>(h2f, ft, Wf, bf,
                                                      (float*)d_out, B, H2, F);
}
